// ClusterPooling_75857712382054
// MI455X (gfx1250) — matrix operand order repaired and checked by reading, not compiled, not run on hardware
//
#include <hip/hip_runtime.h>
#include <hip/hip_bf16.h>
#include <math.h>

typedef __attribute__((ext_vector_type(16))) _Float16 v16h;
typedef __attribute__((ext_vector_type(8)))  float    v8f;

#define CDIM 128  // reference C

// ---------------------------------------------------------------------------
// Init: zero agg + pooled, init labels, zero deg
// ---------------------------------------------------------------------------
__global__ void cp_init_kernel(float* __restrict__ agg, float* __restrict__ pooled,
                               float* __restrict__ deg, int* __restrict__ labels,
                               int n, long long nc) {
    long long i = (long long)blockIdx.x * blockDim.x + threadIdx.x;
    if (i < nc) { agg[i] = 0.0f; pooled[i] = 0.0f; }
    if (i < n)  { deg[i] = 0.0f; labels[i] = (int)i; }
}

// ---------------------------------------------------------------------------
// Projection s1 = x @ w[:C], s2 = x @ w[C:] via V_WMMA_F32_16X16X32_F16.
// One wave32 per 16-node tile; 4 K-chunks of 32 cover C=128.
// blockDim.x == 32, grid.x == ceil(n/16); all WMMA guards are blockIdx-uniform.
// ---------------------------------------------------------------------------
__global__ void cp_proj_wmma_kernel(const float* __restrict__ x, const float* __restrict__ w,
                                    float* __restrict__ s1, float* __restrict__ s2, int n) {
    const int tile = blockIdx.x;
    const int lane = threadIdx.x & 31;
    const int mrow = lane & 15;
    const bool hi  = lane >= 16;

    int node = tile * 16 + mrow;
    int nclamp = node < n ? node : (n - 1);          // clamp loads on tail tile
    const float* xrow = x + (size_t)nclamp * CDIM;

    v8f acc = {};
#pragma unroll
    for (int ch = 0; ch < 4; ++ch) {
        const int kb  = ch * 32 + (hi ? 8  : 0);     // A VGPR0-3 K-run
        const int kb2 = ch * 32 + (hi ? 24 : 16);    // A VGPR4-7 K-run
        v16h a;
#pragma unroll
        for (int j = 0; j < 8; ++j) {
            a[j]     = (_Float16)xrow[kb  + j];
            a[8 + j] = (_Float16)xrow[kb2 + j];
        }
        v16h b;
#pragma unroll
        for (int j = 0; j < 16; ++j) b[j] = (_Float16)0.0f;
        if (mrow < 2) {                               // only N=0 (w1) and N=1 (w2) columns
            const float* wp = w + mrow * CDIM + ch * 32 + (hi ? 8 : 0);
#pragma unroll
            for (int j = 0; j < 16; ++j) b[j] = (_Float16)wp[(j < 8) ? j : (j + 8)];
        }
        acc = __builtin_amdgcn_wmma_f32_16x16x32_f16(
            /*neg_a=*/false, a, /*neg_b=*/false, b,
            /*c_mod=*/(short)0, acc, /*reuse_a=*/false, /*reuse_b=*/false);
    }

    if (mrow < 2) {                                   // lanes 0,1,16,17 hold the results
        float* dst = (mrow == 0) ? s1 : s2;
        const int base = tile * 16 + (hi ? 8 : 0);
#pragma unroll
        for (int r = 0; r < 8; ++r)
            if (base + r < n) dst[base + r] = acc[r];
    }
}

// ---------------------------------------------------------------------------
// Edge scoring: score = tanh(s1[u]+s2[v]+s1[v]+s2[u]+2b); sel = (u!=v)&&score>=0
// Compact edges to int32 + per-edge weight for the L2-resident CC loop.
// ---------------------------------------------------------------------------
__global__ void cp_edge_score_kernel(const long long* __restrict__ ei,
                                     const float* __restrict__ s1, const float* __restrict__ s2,
                                     const float* __restrict__ bptr,
                                     int* __restrict__ u32, int* __restrict__ v32,
                                     int* __restrict__ sel, float* __restrict__ w_e, int E) {
    int i = blockIdx.x * blockDim.x + threadIdx.x;
    if (i >= E) return;
    int u = (int)ei[i];
    int v = (int)ei[(long long)E + i];
    float e = s1[u] + s2[v] + s1[v] + s2[u] + 2.0f * bptr[0];
    float score = tanhf(e);
    int s = (u != v) && (score >= 0.0f);
    u32[i] = u;
    v32[i] = v;
    sel[i] = s;
    w_e[i] = s ? score : 0.0f;
}

// ---------------------------------------------------------------------------
// CC: one label-propagation pass over selected edges (atomic min scatter)
// ---------------------------------------------------------------------------
__global__ void cp_cc_edge_kernel(const int* __restrict__ u32, const int* __restrict__ v32,
                                  const int* __restrict__ sel, int* __restrict__ labels, int E) {
    int i = blockIdx.x * blockDim.x + threadIdx.x;
    if (i >= E) return;
    if (!sel[i]) return;
    int u = u32[i], v = v32[i];
    int lu = labels[u];
    int lv = labels[v];
    if (lv < lu) atomicMin(&labels[u], lv);
    if (lu < lv) atomicMin(&labels[v], lu);
}

// CC: pointer jump dst[i] = src[src[i]] (out-of-place; labels is a shrinking map)
__global__ void cp_cc_jump_kernel(const int* __restrict__ src, int* __restrict__ dst, int n) {
    int i = blockIdx.x * blockDim.x + threadIdx.x;
    if (i >= n) return;
    dst[i] = src[src[i]];
}

// ---------------------------------------------------------------------------
// Edge aggregation: rep = (x[u]+x[v])*w_e scattered to both endpoints (+deg).
// 128 threads per edge (one per channel), 2 edges per 256-thread block.
// ---------------------------------------------------------------------------
__global__ void cp_edge_agg_kernel(const int* __restrict__ u32, const int* __restrict__ v32,
                                   const int* __restrict__ sel, const float* __restrict__ w_e,
                                   const float* __restrict__ x, float* __restrict__ agg,
                                   float* __restrict__ deg, int E) {
    int eidx = blockIdx.x * 2 + (threadIdx.x >> 7);
    int c = threadIdx.x & (CDIM - 1);
    if (eidx >= E) return;
    if (!sel[eidx]) return;
    int u = u32[eidx], v = v32[eidx];
    float we = w_e[eidx];
    float rep = (x[(size_t)u * CDIM + c] + x[(size_t)v * CDIM + c]) * we;
    atomicAdd(&agg[(size_t)u * CDIM + c], rep);
    atomicAdd(&agg[(size_t)v * CDIM + c], rep);
    if (c == 0) {
        atomicAdd(&deg[u], 1.0f);
        atomicAdd(&deg[v], 1.0f);
    }
}

// ---------------------------------------------------------------------------
// node_x = touched ? agg/max(deg,1) : x ; pooled[cluster] += node_x
// ---------------------------------------------------------------------------
__global__ void cp_pool_kernel(const float* __restrict__ x, const float* __restrict__ agg,
                               const float* __restrict__ deg, const int* __restrict__ labels,
                               float* __restrict__ pooled, int n) {
    int node = blockIdx.x;
    int c = threadIdx.x;
    if (node >= n) return;
    float d = deg[node];
    float nx = (d > 0.0f) ? (agg[(size_t)node * CDIM + c] / fmaxf(d, 1.0f))
                          : x[(size_t)node * CDIM + c];
    atomicAdd(&pooled[(size_t)labels[node] * CDIM + c], nx);
}

// ---------------------------------------------------------------------------
// new_edge_index = labels[edge_index] (as f32, exact for n<=1e5); new_batch = 0
// ---------------------------------------------------------------------------
__global__ void cp_remap_kernel(const long long* __restrict__ ei, const int* __restrict__ labels,
                                float* __restrict__ edge_out, float* __restrict__ batch_out,
                                int E2, int n) {
    int i = blockIdx.x * blockDim.x + threadIdx.x;
    if (i < E2) edge_out[i] = (float)labels[(int)ei[i]];
    if (i < n)  batch_out[i] = 0.0f;
}

// ---------------------------------------------------------------------------
extern "C" void kernel_launch(void* const* d_in, const int* in_sizes, int n_in,
                              void* d_out, int out_size, void* d_ws, size_t ws_size,
                              hipStream_t stream) {
    const float*     x  = (const float*)d_in[0];
    const long long* ei = (const long long*)d_in[1];   // int64 [2, E]
    const float*     w  = (const float*)d_in[3];       // [2C]
    const float*     b  = (const float*)d_in[4];       // [1]

    const int N = in_sizes[2];                         // batch has N elements
    const int E = in_sizes[1] / 2;
    const long long NC = (long long)N * CDIM;

    // ---- workspace carve (all 16B-aligned given sizes) ----
    char* ws = (char*)d_ws;
    float* s1      = (float*)ws;                 ws += (size_t)N * 4;
    float* s2      = (float*)ws;                 ws += (size_t)N * 4;
    int*   labelsA = (int*)ws;                   ws += (size_t)N * 4;
    int*   labelsB = (int*)ws;                   ws += (size_t)N * 4;
    float* deg     = (float*)ws;                 ws += (size_t)N * 4;
    float* w_e     = (float*)ws;                 ws += (size_t)E * 4;
    int*   u32     = (int*)ws;                   ws += (size_t)E * 4;
    int*   v32     = (int*)ws;                   ws += (size_t)E * 4;
    int*   sel     = (int*)ws;                   ws += (size_t)E * 4;
    float* agg     = (float*)ws;                 ws += (size_t)NC * 4;

    // ---- output carve (f32 encoding, return order) ----
    float* pooled    = (float*)d_out;            // [N, C]
    float* edge_out  = pooled + NC;              // [2, E]
    float* batch_out = edge_out + (size_t)2 * E; // [N]

    const int T = 256;

    // 0) init
    {
        long long total = NC;
        int blocks = (int)((total + T - 1) / T);
        cp_init_kernel<<<blocks, T, 0, stream>>>(agg, pooled, deg, labelsA, N, NC);
    }

    // 1) projection via WMMA
    {
        int tiles = (N + 15) / 16;
        cp_proj_wmma_kernel<<<tiles, 32, 0, stream>>>(x, w, s1, s2, N);
    }

    // 2) edge scoring + compaction
    cp_edge_score_kernel<<<(E + T - 1) / T, T, 0, stream>>>(ei, s1, s2, b, u32, v32, sel, w_e, E);

    // 3) connected components: 32 x (edge min-prop, two pointer jumps)
    {
        int eb = (E + T - 1) / T;
        int nb = (N + T - 1) / T;
        for (int it = 0; it < 32; ++it) {
            cp_cc_edge_kernel<<<eb, T, 0, stream>>>(u32, v32, sel, labelsA, E);
            cp_cc_jump_kernel<<<nb, T, 0, stream>>>(labelsA, labelsB, N);
            cp_cc_jump_kernel<<<nb, T, 0, stream>>>(labelsB, labelsA, N);
        }
    }

    // 4) weighted edge aggregation (scatter-add to both endpoints)
    cp_edge_agg_kernel<<<(E + 1) / 2, T, 0, stream>>>(u32, v32, sel, w_e, x, agg, deg, E);

    // 5) per-cluster pooled sum
    cp_pool_kernel<<<N, CDIM, 0, stream>>>(x, agg, deg, labelsA, pooled, N);

    // 6) remap edges + zero batch
    {
        int total = 2 * E;
        int blocks = (total + T - 1) / T;
        cp_remap_kernel<<<blocks, T, 0, stream>>>(ei, labelsA, edge_out, batch_out, total, N);
    }
}
